// Attn_Graph_23364622090745
// MI455X (gfx1250) — matrix operand order repaired and checked by reading, not compiled, not run on hardware
//
#include <hip/hip_runtime.h>
#include <hip/hip_bf16.h>
#include <stdint.h>

typedef __attribute__((ext_vector_type(8)))  _Float16 v8h;
typedef __attribute__((ext_vector_type(16))) _Float16 v16h;
typedef __attribute__((ext_vector_type(8)))  float    v8f;

static __device__ inline v16h cat8(v8h lo, v8h hi) {
  return __builtin_shufflevector(lo, hi, 0,1,2,3,4,5,6,7,8,9,10,11,12,13,14,15);
}

#define B_  32
#define C_  256
#define HW  3136   // 56*56
#define KTOP 120
#define KPAD 128
#define LDST 264   // 256 + 8 pad halves to break LDS bank conflicts

// ---------------- weight conversion ----------------
__global__ void k_cvt_h(const float* __restrict__ src, _Float16* __restrict__ dst, int n) {
  int i = blockIdx.x * 256 + threadIdx.x;
  if (i < n) dst[i] = (_Float16)src[i];
}

// dst[c*R + r] = src[r*C + c]  (transpose + f32->f16)
__global__ void k_transpose_h(const float* __restrict__ src, _Float16* __restrict__ dst, int R, int C) {
  int i = blockIdx.x * 256 + threadIdx.x;
  if (i >= R * C) return;
  int r = i / C, c = i % C;
  dst[(size_t)c * R + r] = (_Float16)src[(size_t)r * C + c];
}

// ---------------- conv1: out[b,o,p] = sum_c x[b,c,p] * w1[o,c] + b1[o] ----------------
// M = o (16 tiles), N = pixels (2 tiles of 16 per block), K = 256 channels
__global__ __launch_bounds__(256) void k_conv1(const float* __restrict__ x,
                                               const _Float16* __restrict__ w1h,
                                               const float* __restrict__ c1b,
                                               float* __restrict__ c1) {
  __shared__ __align__(16) _Float16 xs[32 * LDST];
  int blk = blockIdx.x;                 // 32 batches * 98 slabs
  int b = blk / 98, pbase = (blk % 98) * 32;
  int t = threadIdx.x, lane = t & 31, kb = t >> 5;
  // stage x[b, :, pbase..pbase+31] as f16, transposed to [pixel][k]
  const float* xp = x + ((size_t)b * C_) * HW + pbase + lane;
  for (int k = kb; k < 256; k += 8)
    xs[lane * LDST + k] = (_Float16)xp[(size_t)k * HW];
  __syncthreads();

  int w = t >> 5, l = t & 31, half = l >> 4, ln = l & 15;
  v8f acc[2][2] = {};
  for (int kk = 0; kk < 256; kk += 32) {
    v16h a[2];
#pragma unroll
    for (int q = 0; q < 2; ++q) {
      const _Float16* ap = w1h + (size_t)((2 * w + q) * 16 + ln) * 256 + kk;
      a[q] = cat8(*(const v8h*)(ap + half * 8), *(const v8h*)(ap + 16 + half * 8));
    }
    v16h bb[2];
#pragma unroll
    for (int pt = 0; pt < 2; ++pt) {
      const _Float16* bp = &xs[(pt * 16 + ln) * LDST + kk + half * 8];
      bb[pt] = cat8(*(const v8h*)(bp), *(const v8h*)(bp + 16));
    }
#pragma unroll
    for (int q = 0; q < 2; ++q)
#pragma unroll
      for (int pt = 0; pt < 2; ++pt)
        acc[q][pt] = __builtin_amdgcn_wmma_f32_16x16x32_f16(
            false, a[q], false, bb[pt], (short)0, acc[q][pt], false, false);
  }
#pragma unroll
  for (int q = 0; q < 2; ++q)
#pragma unroll
    for (int r = 0; r < 8; ++r) {
      int o = (2 * w + q) * 16 + r + half * 8;
      float bias = c1b[o];
#pragma unroll
      for (int pt = 0; pt < 2; ++pt) {
        int px = pbase + pt * 16 + ln;
        c1[((size_t)b * C_ + o) * HW + px] = acc[q][pt][r] + bias;
      }
    }
}

// ---------------- BN2d stats over (b,h,w) per channel ----------------
__global__ void k_bn2d_stats(const float* __restrict__ c1, const float* __restrict__ g,
                             const float* __restrict__ bta, float* __restrict__ scale,
                             float* __restrict__ shift) {
  __shared__ float s1[256], s2[256];
  int o = blockIdx.x, t = threadIdx.x;
  float a = 0.f, q = 0.f;
  for (int i = t; i < B_ * HW; i += 256) {
    int bb = i / HW, p = i % HW;
    float v = c1[((size_t)bb * C_ + o) * HW + p];
    a += v; q += v * v;
  }
  s1[t] = a; s2[t] = q; __syncthreads();
  for (int s = 128; s > 0; s >>= 1) {
    if (t < s) { s1[t] += s1[t + s]; s2[t] += s2[t + s]; }
    __syncthreads();
  }
  if (t == 0) {
    float n = (float)(B_ * HW);
    float m = s1[0] / n, var = s2[0] / n - m * m;
    float sc = g[o] * rsqrtf(var + 1e-5f);
    scale[o] = sc; shift[o] = bta[o] - m * sc;
  }
}

// ---------------- fused bn+relu -> conv2 + residual + softplus -> prob ----------------
__global__ void k_conv2prob(const float* __restrict__ x, const float* __restrict__ c1,
                            const float* __restrict__ scale, const float* __restrict__ shift,
                            const float* __restrict__ w2, const float* __restrict__ c2b,
                            const float* __restrict__ wd, const float* __restrict__ db,
                            float* __restrict__ prob) {
  int blk = blockIdx.x;
  int b = blk / 13, p = (blk % 13) * 256 + threadIdx.x;
  if (p >= HW) return;
  const float* c1p = c1 + ((size_t)b * C_) * HW + p;
  const float* xp  = x  + ((size_t)b * C_) * HW + p;
  float a2 = 0.f, ad = 0.f;
  for (int o = 0; o < 256; ++o) {
    float r = fmaxf(c1p[(size_t)o * HW] * scale[o] + shift[o], 0.f);
    a2 += r * w2[o];
    ad += xp[(size_t)o * HW] * wd[o];
  }
  float s = a2 + c2b[0] + ad + db[0];
  float sp = fmaxf(s, 0.f) + log1pf(expf(-fabsf(s)));   // stable softplus
  prob[b * HW + p] = sp;
}

// ---------------- top-K per batch (monotone: top-K of prob == top-K of softmax) ----------------
__global__ void k_topk(const float* __restrict__ prob, int* __restrict__ indWs,
                       int* __restrict__ indOut) {
  __shared__ float vals[HW];
  __shared__ float sv[256];
  __shared__ int   si[256];
  int b = blockIdx.x, t = threadIdx.x;
  for (int i = t; i < HW; i += 256) vals[i] = prob[b * HW + i];
  __syncthreads();
  for (int it = 0; it < KTOP; ++it) {
    float bv = -INFINITY; int bi = 0x7fffffff;
    for (int i = t; i < HW; i += 256) {
      float v = vals[i];
      if (v > bv || (v == bv && i < bi)) { bv = v; bi = i; }
    }
    sv[t] = bv; si[t] = bi; __syncthreads();
    for (int s = 128; s > 0; s >>= 1) {
      if (t < s) {
        float v = sv[t + s]; int j = si[t + s];
        if (v > sv[t] || (v == sv[t] && j < si[t])) { sv[t] = v; si[t] = j; }
      }
      __syncthreads();
    }
    if (t == 0) {
      int wnr = si[0];
      indWs[b * KTOP + it] = wnr;
      indOut[b * KTOP + it] = wnr;
      vals[wnr] = -INFINITY;
    }
    __syncthreads();
  }
}

// ---------------- gather node features: feat[b,c,k] = relu(x)*prob at top-k, K padded to 128 ----
__global__ void k_feat(const float* __restrict__ x, const float* __restrict__ prob,
                       const int* __restrict__ ind, _Float16* __restrict__ featH) {
  int blk = blockIdx.x;
  int b = blk >> 8, c = blk & 255, t = threadIdx.x;   // 128 threads
  float v = 0.f;
  if (t < KTOP) {
    int p = ind[b * KTOP + t];
    float xv = fmaxf(x[((size_t)b * C_ + c) * HW + p], 0.f);
    v = xv * prob[b * HW + p];
  }
  featH[((size_t)b * C_ + c) * KPAD + t] = (_Float16)v;
}

// ---------------- adjacency: dist -> symmetric degree norm, f16, zero-padded to 128 ----------------
__global__ void k_adj(const int* __restrict__ ind, _Float16* __restrict__ adjH) {
  __shared__ float fx[KTOP], fy[KTOP], dd[KPAD];
  int b = blockIdx.x, t = threadIdx.x;   // 128 threads
  if (t < KTOP) {
    int p = ind[b * KTOP + t];
    fx[t] = (float)(p / 56);
    fy[t] = (float)(p % 56);
  }
  __syncthreads();
  float di = 0.f;
  if (t < KTOP) {
    float s = 0.f;
    for (int j = 0; j < KTOP; ++j) {
      float dx = fx[t] - fx[j], dy = fy[t] - fy[j];
      s += sqrtf(dx * dx + dy * dy);
    }
    di = rsqrtf(s);
  }
  dd[t] = (t < KTOP) ? di : 0.f;
  __syncthreads();
  _Float16* out = adjH + (size_t)b * KPAD * KPAD;
  for (int j = 0; j < KPAD; ++j) {
    float v = 0.f;
    if (t < KTOP && j < KTOP) {
      float dx = fx[t] - fx[j], dy = fy[t] - fy[j];
      v = sqrtf(dx * dx + dy * dy) * dd[t] * dd[j];
    }
    out[t * KPAD + j] = (_Float16)v;
  }
}

// ---------------- gemmA: Tt[b,i,c] = sum_j X[b,c,j] * Adj[b,i,j]  (K=128, 4 WMMA) -------------
// NOTE: output stored TRANSPOSED ([b][i][c]) so gemmW's B-fragments are contiguous row loads.
__global__ __launch_bounds__(256) void k_gemmA(const _Float16* __restrict__ X,
                                               const _Float16* __restrict__ Adj,
                                               _Float16* __restrict__ Tt, int C) {
  int blk = blockIdx.x, mtpb = C >> 4;
  int b = blk / mtpb, mt = blk % mtpb;
  int t = threadIdx.x, w = t >> 5, l = t & 31, half = l >> 4, ln = l & 15;
  const _Float16* xrow = X + ((size_t)b * C + mt * 16 + ln) * KPAD;
  const _Float16* brow = Adj + ((size_t)b * KPAD + w * 16 + ln) * KPAD;
  v8f acc = {};
#pragma unroll
  for (int kk = 0; kk < KPAD; kk += 32) {
    v16h a = cat8(*(const v8h*)(xrow + kk + half * 8),
                  *(const v8h*)(xrow + kk + 16 + half * 8));
    v16h bb = cat8(*(const v8h*)(brow + kk + half * 8),
                   *(const v8h*)(brow + kk + 16 + half * 8));
    acc = __builtin_amdgcn_wmma_f32_16x16x32_f16(false, a, false, bb, (short)0, acc, false, false);
  }
  int i = w * 16 + ln;
  _Float16* orow = Tt + ((size_t)b * KPAD + i) * C + mt * 16;
#pragma unroll
  for (int r = 0; r < 8; ++r)
    orow[r + half * 8] = (_Float16)acc[r];
}

// ---------------- gemmW: Y[b,o,i] = sum_c Wt[o,c] * Tt[b,i,c] ----------------
// Cin compile-time so the K-loop fully unrolls and loads get clause-grouped.
template <int CIN>
__global__ __launch_bounds__(256) void k_gemmW(const _Float16* __restrict__ Wt,
                                               const _Float16* __restrict__ Tt,
                                               float* __restrict__ outF,
                                               _Float16* __restrict__ outH,
                                               int O, int relu) {
  int blk = blockIdx.x, mtpb = O >> 4;
  int b = blk / mtpb, mt = blk % mtpb;
  int t = threadIdx.x, w = t >> 5, l = t & 31, half = l >> 4, ln = l & 15;
  const _Float16* arow = Wt + (size_t)(mt * 16 + ln) * CIN;
  const _Float16* brow = Tt + ((size_t)b * KPAD + w * 16 + ln) * CIN;  // row i, contiguous in c
  v8f acc = {};
#pragma unroll
  for (int kk = 0; kk < CIN; kk += 32) {
    v16h a = cat8(*(const v8h*)(arow + kk + half * 8),
                  *(const v8h*)(arow + kk + 16 + half * 8));
    v16h bb = cat8(*(const v8h*)(brow + kk + half * 8),
                   *(const v8h*)(brow + kk + 16 + half * 8));
    acc = __builtin_amdgcn_wmma_f32_16x16x32_f16(false, a, false, bb, (short)0, acc, false, false);
  }
#pragma unroll
  for (int r = 0; r < 8; ++r) {
    int o = mt * 16 + r + half * 8, i = w * 16 + ln;
    float v = acc[r];
    if (relu) v = fmaxf(v, 0.f);
    size_t idx = ((size_t)b * O + o) * KPAD + i;
    if (outF) outF[idx] = v;
    if (outH) outH[idx] = (_Float16)v;
  }
}

// ---------------- BN1d stats over (b, k<120) per channel ----------------
__global__ void k_bn1d_stats(const float* __restrict__ y, const float* __restrict__ g,
                             const float* __restrict__ bta, float* __restrict__ scale,
                             float* __restrict__ shift) {
  __shared__ float s1[128], s2[128];
  int o = blockIdx.x, t = threadIdx.x;   // 128 threads
  float a = 0.f, q = 0.f;
  for (int i = t; i < B_ * KTOP; i += 128) {
    int bb = i / KTOP, k = i % KTOP;
    float v = y[((size_t)bb * C_ + o) * KPAD + k];
    a += v; q += v * v;
  }
  s1[t] = a; s2[t] = q; __syncthreads();
  for (int s = 64; s > 0; s >>= 1) {
    if (t < s) { s1[t] += s1[t + s]; s2[t] += s2[t + s]; }
    __syncthreads();
  }
  if (t == 0) {
    float n = (float)(B_ * KTOP);
    float m = s1[0] / n, var = s2[0] / n - m * m;
    float sc = g[o] * rsqrtf(var + 1e-5f);
    scale[o] = sc; shift[o] = bta[o] - m * sc;
  }
}

__global__ void k_bn1d_apply(const float* __restrict__ y, const float* __restrict__ scale,
                             const float* __restrict__ shift, _Float16* __restrict__ out) {
  int blk = blockIdx.x;
  int b = blk >> 8, o = blk & 255, i = threadIdx.x;   // 128 threads
  float v = 0.f;
  if (i < KTOP) v = fmaxf(y[((size_t)b * C_ + o) * KPAD + i] * scale[o] + shift[o], 0.f);
  out[((size_t)b * C_ + o) * KPAD + i] = (_Float16)v;
}

// ---------------- head: mean over k then 64->32->10 MLP ----------------
__global__ void k_head(const float* __restrict__ y4, const float* __restrict__ m1w,
                       const float* __restrict__ m1b, const float* __restrict__ m2w,
                       const float* __restrict__ m2b, float* __restrict__ out) {
  __shared__ float m[64], h[32];
  int b = blockIdx.x, t = threadIdx.x;   // 64 threads
  float s = 0.f;
  for (int i = 0; i < KTOP; ++i) s += y4[((size_t)b * 64 + t) * KPAD + i];
  m[t] = s / (float)KTOP;
  __syncthreads();
  if (t < 32) {
    float a = m1b[t];
    for (int c = 0; c < 64; ++c) a += m[c] * m1w[c * 32 + t];
    h[t] = a;
  }
  __syncthreads();
  if (t < 10) {
    float a = m2b[t];
    for (int c = 0; c < 32; ++c) a += h[c] * m2w[c * 10 + t];
    out[b * 10 + t] = a;
  }
}

extern "C" void kernel_launch(void* const* d_in, const int* in_sizes, int n_in,
                              void* d_out, int out_size, void* d_ws, size_t ws_size,
                              hipStream_t stream) {
  const float* x      = (const float*)d_in[0];
  const float* c1w    = (const float*)d_in[1];
  const float* c1b    = (const float*)d_in[2];
  const float* c2w    = (const float*)d_in[3];
  const float* c2b    = (const float*)d_in[4];
  const float* dw     = (const float*)d_in[5];
  const float* dbi    = (const float*)d_in[6];
  const float* bn1g   = (const float*)d_in[7];
  const float* bn1b   = (const float*)d_in[8];
  const float* bn2g   = (const float*)d_in[9];
  const float* bn2b   = (const float*)d_in[10];
  const float* g1     = (const float*)d_in[11];
  const float* g2     = (const float*)d_in[12];
  const float* g3     = (const float*)d_in[13];
  const float* g4     = (const float*)d_in[14];
  const float* m1w    = (const float*)d_in[15];
  const float* m1b    = (const float*)d_in[16];
  const float* m2w    = (const float*)d_in[17];
  const float* m2b    = (const float*)d_in[18];

  uint8_t* p = (uint8_t*)d_ws;
  auto carve = [&](size_t bytes) -> void* {
    void* r = (void*)p;
    p += (bytes + 255) & ~(size_t)255;
    return r;
  };
  _Float16* w1h   = (_Float16*)carve(65536 * 2);
  _Float16* g1t   = (_Float16*)carve(65536 * 2);
  _Float16* g2t   = (_Float16*)carve(32768 * 2);
  _Float16* g3t   = (_Float16*)carve(16384 * 2);
  _Float16* g4t   = (_Float16*)carve(8192 * 2);
  float*    c1o   = (float*)carve((size_t)B_ * C_ * HW * 4);
  float*    sc1   = (float*)carve(256 * 4);
  float*    sh1   = (float*)carve(256 * 4);
  float*    prob  = (float*)carve((size_t)B_ * HW * 4);
  int*      indW  = (int*)carve((size_t)B_ * KTOP * 4);
  _Float16* featH = (_Float16*)carve((size_t)B_ * C_ * KPAD * 2);
  _Float16* adjH  = (_Float16*)carve((size_t)B_ * KPAD * KPAD * 2);
  _Float16* Tbuf  = (_Float16*)carve((size_t)B_ * KPAD * C_ * 2);
  float*    y1f   = (float*)carve((size_t)B_ * C_ * KPAD * 4);
  _Float16* y1h   = (_Float16*)carve((size_t)B_ * C_ * KPAD * 2);
  _Float16* y2h   = (_Float16*)carve((size_t)B_ * 128 * KPAD * 2);
  _Float16* y3h   = (_Float16*)carve((size_t)B_ * 128 * KPAD * 2);
  float*    y4f   = (float*)carve((size_t)B_ * 64 * KPAD * 4);
  float*    sc2   = (float*)carve(256 * 4);
  float*    sh2   = (float*)carve(256 * 4);

  float* y6out  = (float*)d_out;
  int*   indOut = ((int*)d_out) + B_ * 10;   // 320 floats, then 3840 int32 (bitwise)

  // weight prep (f32 -> f16, GCN weights transposed to [o][c])
  k_cvt_h<<<(65536 + 255) / 256, 256, 0, stream>>>(c1w, w1h, 65536);
  k_transpose_h<<<(65536 + 255) / 256, 256, 0, stream>>>(g1, g1t, 256, 256);
  k_transpose_h<<<(32768 + 255) / 256, 256, 0, stream>>>(g2, g2t, 256, 128);
  k_transpose_h<<<(16384 + 255) / 256, 256, 0, stream>>>(g3, g3t, 128, 128);
  k_transpose_h<<<(8192 + 255) / 256, 256, 0, stream>>>(g4, g4t, 128, 64);

  // conv1 GEMM (WMMA f16)
  k_conv1<<<B_ * (HW / 32), 256, 0, stream>>>(x, w1h, c1b, c1o);
  // BN2d stats
  k_bn2d_stats<<<256, 256, 0, stream>>>(c1o, bn1g, bn1b, sc1, sh1);
  // fused bn+relu -> conv2 + residual + softplus
  k_conv2prob<<<B_ * 13, 256, 0, stream>>>(x, c1o, sc1, sh1, c2w, c2b, dw, dbi, prob);
  // top-K
  k_topk<<<B_, 256, 0, stream>>>(prob, indW, indOut);
  // gather features + adjacency
  k_feat<<<B_ * 256, 128, 0, stream>>>(x, prob, indW, featH);
  k_adj<<<B_, 128, 0, stream>>>(indW, adjH);

  // GCN layer 1: 256 -> 256, then BN1d + relu
  k_gemmA<<<B_ * 16, 256, 0, stream>>>(featH, adjH, Tbuf, 256);
  k_gemmW<256><<<B_ * 16, 256, 0, stream>>>(g1t, Tbuf, y1f, (_Float16*)nullptr, 256, 0);
  k_bn1d_stats<<<256, 128, 0, stream>>>(y1f, bn2g, bn2b, sc2, sh2);
  k_bn1d_apply<<<B_ * 256, 128, 0, stream>>>(y1f, sc2, sh2, y1h);
  // GCN layer 2: 256 -> 128, relu
  k_gemmA<<<B_ * 16, 256, 0, stream>>>(y1h, adjH, Tbuf, 256);
  k_gemmW<256><<<B_ * 8, 256, 0, stream>>>(g2t, Tbuf, (float*)nullptr, y2h, 128, 1);
  // GCN layer 3: 128 -> 128, relu
  k_gemmA<<<B_ * 8, 256, 0, stream>>>(y2h, adjH, Tbuf, 128);
  k_gemmW<128><<<B_ * 8, 256, 0, stream>>>(g3t, Tbuf, (float*)nullptr, y3h, 128, 1);
  // GCN layer 4: 128 -> 64, no relu
  k_gemmA<<<B_ * 8, 256, 0, stream>>>(y3h, adjH, Tbuf, 128);
  k_gemmW<128><<<B_ * 4, 256, 0, stream>>>(g4t, Tbuf, y4f, (_Float16*)nullptr, 64, 0);

  // mean over K + MLP head
  k_head<<<B_, 64, 0, stream>>>(y4f, m1w, m1b, m2w, m2b, y6out);

  (void)in_sizes; (void)n_in; (void)out_size; (void)ws_size;
}
